// DSQGAttentionN_82729660056320
// MI455X (gfx1250) — matrix operand order repaired and checked by reading, not compiled, not run on hardware
//
#include <hip/hip_runtime.h>
#include <hip/hip_bf16.h>

// ---------------------------------------------------------------------------
// DSQG attention for MI455X (gfx1250, wave32, WMMA + async-to-LDS pipeline).
// Pipeline: bf16 convert -> WMMA GEMM (qkv) + WMMA GEMM (gate)
//           -> banded attention w/ online softmax + gate (VALU + shfl)
//           -> WMMA GEMM (out proj).
// GEMMs use GLOBAL_LOAD_ASYNC_TO_LDS_B128 double buffering (ASYNCcnt) and
// ds_load_b128 fragment reads feeding v_wmma_f32_16x16x32_bf16.
// ---------------------------------------------------------------------------

typedef __attribute__((ext_vector_type(16))) __bf16 v16bf;
typedef __attribute__((ext_vector_type(8)))  float  v8f;

#define SEQ_N     2048
#define DIM_D     1024
#define NUM_HEADS 16
#define HEAD_DIM  64
#define NUM_OFF   44

__device__ __constant__ int c_offs[NUM_OFF] = {
    0,1,2,3,4,5,6,7,8,9,10,11,12,13,14,15,16,17,18,19,20,21,22,23,24,25,
    26,27,28,29,30,31,32,48,64,96,128,192,256,384,512,768,1024,1536};

// ---- f32 -> bf16 (round-to-nearest-even) ----------------------------------
__device__ __forceinline__ unsigned short f32_to_bf16(float f) {
    unsigned int u = __float_as_uint(f);
    u += 0x7fffu + ((u >> 16) & 1u);
    return (unsigned short)(u >> 16);
}

__global__ void cvt_f32_bf16_kernel(const float* __restrict__ src,
                                    unsigned short* __restrict__ dst, int n) {
    int i = blockIdx.x * blockDim.x + threadIdx.x;
    if (i < n) dst[i] = f32_to_bf16(src[i]);
}

// ---- WMMA bf16 GEMM: C[M,N] = A[M,K] * B[N,K]^T + bias[N] -----------------
// Block: 256 threads = 8 waves (4 M-waves x 2 N-waves).
// Block tile 128(M) x 64(N); wave tile 32x32; K staged 64-wide, double
// buffered in LDS via async global->LDS copies.
// Requires M%128==0, N%64==0, K%64==0 (true for all three GEMMs here).

#define KSTAGE      64
#define LDS_STRIDE  144                       // 64 bf16 = 128B, +16B pad
#define A_TILE_B    (128 * LDS_STRIDE)        // 18432 B
#define B_TILE_B    (64 * LDS_STRIDE)         //  9216 B
#define STAGE_B     (A_TILE_B + B_TILE_B)     // 27648 B

union FragBF { v16bf v; uint4 u[2]; };

__device__ __forceinline__ void async_cp16(unsigned lds_addr,
                                           const unsigned short* gptr) {
    unsigned long long ga = (unsigned long long)gptr;
    asm volatile("global_load_async_to_lds_b128 %0, %1, off"
                 :: "v"(lds_addr), "v"(ga) : "memory");
}

__global__ __launch_bounds__(256)
void gemm_bf16_wmma_kernel(const unsigned short* __restrict__ A,   // [M,K]
                           const unsigned short* __restrict__ B,   // [N,K]
                           const float* __restrict__ bias,          // [N]
                           float* __restrict__ C,                   // [M,N]
                           int M, int N, int K) {
    __shared__ alignas(16) unsigned char lds[2 * STAGE_B];

    const int tid   = threadIdx.x;
    const int lane  = tid & 31;
    const int wave  = tid >> 5;
    const int r     = lane & 15;    // row (A) / col (B,C) within 16
    const int half  = lane >> 4;    // lane group 0/1
    const int waveM = wave >> 1;    // 0..3
    const int waveN = wave & 1;     // 0..1

    const int rowBase = blockIdx.y * 128;
    const int colBase = blockIdx.x * 64;

    const unsigned ldsBase = (unsigned)(unsigned long long)(&lds[0]);

    // Issue one stage of async copies: A 128x64 (1024 chunks) + B 64x64
    // (512 chunks) of 16B; 6 chunks per thread. Ascending issue order ->
    // in-order completion -> s_wait_asynccnt 6 retires exactly one stage.
    auto issue_stage = [&](int k0, int buf) {
        const unsigned base = ldsBase + (unsigned)buf * STAGE_B;
#pragma unroll
        for (int i = 0; i < 4; ++i) {
            const int c = tid + i * 256;
            const int row = c >> 3, ch = c & 7;
            async_cp16(base + row * LDS_STRIDE + ch * 16,
                       A + (size_t)(rowBase + row) * K + k0 + ch * 8);
        }
#pragma unroll
        for (int i = 0; i < 2; ++i) {
            const int c = tid + i * 256;
            const int row = c >> 3, ch = c & 7;
            async_cp16(base + A_TILE_B + row * LDS_STRIDE + ch * 16,
                       B + (size_t)(colBase + row) * K + k0 + ch * 8);
        }
    };

    v8f acc[2][2];
#pragma unroll
    for (int mt = 0; mt < 2; ++mt)
#pragma unroll
        for (int nt = 0; nt < 2; ++nt)
            acc[mt][nt] = (v8f){0.f, 0.f, 0.f, 0.f, 0.f, 0.f, 0.f, 0.f};

    const int nstages = K / KSTAGE;
    issue_stage(0, 0);

    for (int s = 0; s < nstages; ++s) {
        if (s + 1 < nstages) {
            issue_stage((s + 1) * KSTAGE, (s + 1) & 1);
            asm volatile("s_wait_asynccnt 0x6" ::: "memory");  // stage s done
        } else {
            asm volatile("s_wait_asynccnt 0x0" ::: "memory");
        }
        __syncthreads();

        const unsigned char* Ab = lds + (s & 1) * STAGE_B;
        const unsigned char* Bb = Ab + A_TILE_B;

#pragma unroll
        for (int ks = 0; ks < 2; ++ks) {
            // A fragments (ISA 16-bit A 16x32 layout): lane holds bytes
            // [ks*64 + half*16, +16) and [ks*64 + 32 + half*16, +16) of row r.
            FragBF a[2];
#pragma unroll
            for (int mt = 0; mt < 2; ++mt) {
                const unsigned char* ap =
                    Ab + (waveM * 32 + mt * 16 + r) * LDS_STRIDE +
                    ks * 64 + half * 16;
                a[mt].u[0] = *(const uint4*)(ap);
                a[mt].u[1] = *(const uint4*)(ap + 32);
            }
#pragma unroll
            for (int nt = 0; nt < 2; ++nt) {
                // B fragment: lane (col = r) holds bytes
                // [ks*64 + half*16, +16) and [ks*64 + 32 + half*16, +16).
                const unsigned char* bp =
                    Bb + (waveN * 32 + nt * 16 + r) * LDS_STRIDE +
                    ks * 64 + half * 16;
                FragBF b;
                b.u[0] = *(const uint4*)(bp);
                b.u[1] = *(const uint4*)(bp + 32);
                acc[0][nt] = __builtin_amdgcn_wmma_f32_16x16x32_bf16(
                    false, a[0].v, false, b.v, (short)0, acc[0][nt], false,
                    false);
                acc[1][nt] = __builtin_amdgcn_wmma_f32_16x16x32_bf16(
                    false, a[1].v, false, b.v, (short)0, acc[1][nt], false,
                    false);
            }
        }
        __syncthreads();
    }

    // C/D layout: lane (col = r), VGPR i -> row = half*8 + i
#pragma unroll
    for (int mt = 0; mt < 2; ++mt) {
#pragma unroll
        for (int nt = 0; nt < 2; ++nt) {
            const int col = colBase + waveN * 32 + nt * 16 + r;
            const float bv = bias[col];
#pragma unroll
            for (int i = 0; i < 8; ++i) {
                const int row = rowBase + waveM * 32 + mt * 16 + half * 8 + i;
                C[(size_t)row * N + col] = acc[mt][nt][i] + bv;
            }
        }
    }
}

// ---- Banded attention + sigmoid gate --------------------------------------
// One wave32 per (n, h). Lane owns dims {2*lane, 2*lane+1} of HEAD_DIM=64.
// Online softmax over sorted offsets; causal break once n < offset.
__global__ __launch_bounds__(256)
void attn_gate_kernel(const float* __restrict__ qkv,     // [N, 3*D]
                      const float* __restrict__ gatep,   // [N, D] (pre-sigmoid)
                      const float* __restrict__ pos_bias,// [NUM_OFF, H]
                      unsigned short* __restrict__ ab,   // [N, D] bf16 output
                      int N) {
    const int gtid = blockIdx.x * blockDim.x + threadIdx.x;
    const int wid  = gtid >> 5;
    const int lane = gtid & 31;
    const int n = wid >> 4;          // consecutive waves share n (k/v reuse)
    const int h = wid & 15;
    if (n >= N) return;

    const float2 q2 =
        ((const float2*)(qkv + (size_t)n * (3 * DIM_D) + h * HEAD_DIM))[lane];

    float m = -__builtin_inff();
    float l = 0.f, acc0 = 0.f, acc1 = 0.f;

#pragma unroll 1
    for (int o = 0; o < NUM_OFF; ++o) {
        const int off = c_offs[o];
        if (n < off) break;  // offsets ascending: all later ones masked too
        const float* krow =
            qkv + (size_t)(n - off) * (3 * DIM_D) + DIM_D + h * HEAD_DIM;
        const float2 k2 = ((const float2*)krow)[lane];
        const float2 v2 = ((const float2*)(krow + DIM_D))[lane];

        float p = q2.x * k2.x + q2.y * k2.y;
        p += __shfl_xor(p, 16, 32);
        p += __shfl_xor(p, 8, 32);
        p += __shfl_xor(p, 4, 32);
        p += __shfl_xor(p, 2, 32);
        p += __shfl_xor(p, 1, 32);

        const float s  = p * 0.125f + pos_bias[o * NUM_HEADS + h];
        const float mn = fmaxf(m, s);
        const float cs = __expf(m - mn);   // 0 on first iteration (m = -inf)
        const float w  = __expf(s - mn);
        l    = l * cs + w;
        acc0 = acc0 * cs + w * v2.x;
        acc1 = acc1 * cs + w * v2.y;
        m = mn;
    }

    const float inv = 1.f / l;  // offset 0 always valid -> l > 0
    const float2 g2 =
        ((const float2*)(gatep + (size_t)n * DIM_D + h * HEAD_DIM))[lane];
    const float g0 = 1.f / (1.f + __expf(-g2.x));
    const float g1 = 1.f / (1.f + __expf(-g2.y));
    const unsigned int u0 = f32_to_bf16(acc0 * inv * g0);
    const unsigned int u1 = f32_to_bf16(acc1 * inv * g1);
    *(unsigned int*)(ab + (size_t)n * DIM_D + h * HEAD_DIM + lane * 2) =
        u0 | (u1 << 16);
}

// ---------------------------------------------------------------------------
extern "C" void kernel_launch(void* const* d_in, const int* in_sizes, int n_in,
                              void* d_out, int out_size, void* d_ws,
                              size_t ws_size, hipStream_t stream) {
    const float* x     = (const float*)d_in[0];  // [2048,1024]
    const float* Wqkv  = (const float*)d_in[1];  // [3072,1024]
    const float* bqkv  = (const float*)d_in[2];  // [3072]
    const float* Wout  = (const float*)d_in[3];  // [1024,1024]
    const float* bout  = (const float*)d_in[4];  // [1024]
    const float* Wgate = (const float*)d_in[5];  // [1024,1024]
    const float* bgate = (const float*)d_in[6];  // [1024]
    const float* pbias = (const float*)d_in[7];  // [44,16]
    float* out = (float*)d_out;

    const int N = SEQ_N, D = DIM_D;

    char* ws = (char*)d_ws;
    size_t off = 0;
    auto alloc = [&](size_t bytes) -> void* {
        void* p = ws + off;
        off = (off + bytes + 255) & ~(size_t)255;
        return p;
    };
    unsigned short* xb     = (unsigned short*)alloc((size_t)N * D * 2);
    unsigned short* wqkvb  = (unsigned short*)alloc((size_t)3 * D * D * 2);
    unsigned short* wgateb = (unsigned short*)alloc((size_t)D * D * 2);
    unsigned short* woutb  = (unsigned short*)alloc((size_t)D * D * 2);
    float*          qkvf   = (float*)alloc((size_t)N * 3 * D * 4);
    float*          gatef  = (float*)alloc((size_t)N * D * 4);
    unsigned short* ab     = (unsigned short*)alloc((size_t)N * D * 2);

    // 1) convert fp32 -> bf16
    auto cvt = [&](const float* s, unsigned short* d, int n) {
        cvt_f32_bf16_kernel<<<(n + 255) / 256, 256, 0, stream>>>(s, d, n);
    };
    cvt(x,     xb,     N * D);
    cvt(Wqkv,  wqkvb,  3 * D * D);
    cvt(Wgate, wgateb, D * D);
    cvt(Wout,  woutb,  D * D);

    // 2) qkv = x @ Wqkv^T + bqkv    (M=2048, N=3072, K=1024)
    gemm_bf16_wmma_kernel<<<dim3(3 * D / 64, N / 128), 256, 0, stream>>>(
        xb, wqkvb, bqkv, qkvf, N, 3 * D, D);
    //    gate_pre = x @ Wgate^T + bgate  (M=2048, N=1024, K=1024)
    gemm_bf16_wmma_kernel<<<dim3(D / 64, N / 128), 256, 0, stream>>>(
        xb, wgateb, bgate, gatef, N, D, D);

    // 3) banded attention + sigmoid gate -> bf16 activations
    {
        const int waves = N * NUM_HEADS;          // 32768 wave32's
        const int threads = waves * 32;
        attn_gate_kernel<<<threads / 256, 256, 0, stream>>>(qkvf, gatef, pbias,
                                                            ab, N);
    }

    // 4) out = (attn * gate) @ Wout^T + bout  (M=2048, N=1024, K=1024)
    gemm_bf16_wmma_kernel<<<dim3(D / 64, N / 128), 256, 0, stream>>>(
        ab, woutb, bout, out, N, D, D);
}
